// MultiHeadAttention_27925877358799
// MI455X (gfx1250) — matrix operand order repaired and checked by reading, not compiled, not run on hardware
//
#include <hip/hip_runtime.h>
#include <hip/hip_bf16.h>

// ---------- MHA forward on gfx1250 via bf16 WMMA (v_wmma_f32_16x16x32_bf16) ----
// D_MODEL=1024, N_HEADS=16, HEAD_DIM=64, B=4, T=2048, M = B*T = 8192.
// Projections use the 3xBF16 split scheme (hi+lo) for ~fp32-class accuracy;
// the flash-attention core runs in plain bf16 with fp32 accumulation.

#define D_MODEL 1024
#define N_HEADS 16
#define HEAD_DIM 64
#define SEQ_T 2048
#define BATCH 4
#define M_TOT (BATCH * SEQ_T)  // 8192

typedef __attribute__((ext_vector_type(16))) __bf16 v16bf;
typedef __attribute__((ext_vector_type(8)))  __bf16 v8bf;
typedef __attribute__((ext_vector_type(8)))  float  v8f;

__device__ __forceinline__ v8f bf_wmma(v16bf a, v16bf b, v8f c) {
  return __builtin_amdgcn_wmma_f32_16x16x32_bf16(false, a, false, b,
                                                 (short)0, c, false, false);
}

__device__ __forceinline__ v16bf cat8(v8bf lo, v8bf hi) {
  return __builtin_shufflevector(lo, hi, 0, 1, 2, 3, 4, 5, 6, 7,
                                 8, 9, 10, 11, 12, 13, 14, 15);
}

// A fragment 16x32 (M x K), row-major source, row stride ld (elements).
// lanes 0-15: row=row0+lane, K {k..k+7, k+16..k+23}
// lanes 16-31: row=row0+lane-16, K {k+8..k+15, k+24..k+31}
__device__ __forceinline__ v16bf load_a(const __bf16* base, int row0, int ld,
                                        int k, int lane) {
  int l = lane & 15, hi = lane >> 4;
  const __bf16* p = base + (size_t)(row0 + l) * ld + k + hi * 8;
  v8bf x0 = *(const v8bf*)p;
  v8bf x1 = *(const v8bf*)(p + 16);
  return cat8(x0, x1);
}

// B fragment 32x16 (K x N). Source is "transposed": row r of `base` holds the
// K-run for output column n = row0 + r.  lanes 0-15: K {k..k+7, k+16..k+23};
// lanes 16-31: K {k+8..k+15, k+24..k+31}.
__device__ __forceinline__ v16bf load_b(const __bf16* base, int row0, int ld,
                                        int k, int lane) {
  int l = lane & 15, hi = lane >> 4;
  const __bf16* p = base + (size_t)(row0 + l) * ld + k + hi * 8;
  v8bf x0 = *(const v8bf*)p;
  v8bf x1 = *(const v8bf*)(p + 16);
  return cat8(x0, x1);
}

// ---------------- prep kernels ----------------------------------------------

// fp32 -> (bf16 hi, bf16 lo) split: v ~= hi + lo with ~16-bit mantissa total.
__global__ __launch_bounds__(256) void cast_split_bf16(
    const float* __restrict__ in, __bf16* __restrict__ hi,
    __bf16* __restrict__ lo, int n) {
  int i = (blockIdx.x * 256 + threadIdx.x) * 4;
  if (i + 3 < n) {
    float4 f = *(const float4*)(in + i);
    float v[4] = {f.x, f.y, f.z, f.w};
#pragma unroll
    for (int j = 0; j < 4; ++j) {
      __bf16 h = (__bf16)v[j];
      hi[i + j] = h;
      lo[i + j] = (__bf16)(v[j] - (float)h);
    }
  }
}

// W [1024][1024] fp32 (W[k][n]) -> Wt_hi/Wt_lo [1024][1024] bf16 (Wt[n][k])
__global__ __launch_bounds__(256) void transpose_split_w(
    const float* __restrict__ W, __bf16* __restrict__ Wh,
    __bf16* __restrict__ Wl) {
  __shared__ float tile[32][33];
  int bx = blockIdx.x & 31;   // n-tile
  int by = blockIdx.x >> 5;   // k-tile
  int tx = threadIdx.x & 31;
  int ty = threadIdx.x >> 5;  // 0..7
#pragma unroll
  for (int j = 0; j < 32; j += 8)
    tile[ty + j][tx] = W[(size_t)(by * 32 + ty + j) * D_MODEL + bx * 32 + tx];
  __syncthreads();
#pragma unroll
  for (int j = 0; j < 32; j += 8) {
    float v = tile[tx][ty + j];
    __bf16 h = (__bf16)v;
    size_t idx = (size_t)(bx * 32 + ty + j) * D_MODEL + by * 32 + tx;
    Wh[idx] = h;
    Wl[idx] = (__bf16)(v - (float)h);
  }
}

// ---------------- split-precision WMMA GEMM ---------------------------------
// Y[m][n] = sum_k (Ah+Al)[m][k] * (Wh+Wl)[n][k] + bias[n]   (drops lo*lo term)
// Block = 256 threads = 8 waves; wave tile = 16(M) x 64(N); block = 128M x 64N.
// mode 0: store bf16 head-split [BH][T][64]             (Q, K)
// mode 2: store bf16 head-split transposed [BH][64][T]  (V)
// mode 3: store fp32 [M][1024]                          (final output)
__global__ __launch_bounds__(256) void gemm_split_wmma(
    const __bf16* __restrict__ Ah, const __bf16* __restrict__ Al,
    const __bf16* __restrict__ Wh, const __bf16* __restrict__ Wl,
    const float* __restrict__ bias, void* __restrict__ outp, int M, int Kd,
    int mode) {
  int lane = threadIdx.x & 31;
  int wave = threadIdx.x >> 5;
  int mBlocks = M >> 7;
  int bm = blockIdx.x % mBlocks;
  int bn = blockIdx.x / mBlocks;
  int m0 = bm * 128 + wave * 16;
  int n0 = bn * 64;
  int l = lane & 15, hi = lane >> 4;

  v8f c[4];
#pragma unroll
  for (int nt = 0; nt < 4; ++nt) c[nt] = (v8f)(0.0f);

  for (int k = 0; k < Kd; k += 32) {
    if (k + 64 < Kd) {  // hide L2 latency for the next-next k-step
      __builtin_prefetch(Ah + (size_t)(m0 + l) * Kd + k + 64, 0, 3);
      __builtin_prefetch(Wh + (size_t)(n0 + l) * Kd + k + 64, 0, 3);
    }
    v16bf ah = load_a(Ah, m0, Kd, k, lane);
    v16bf al = load_a(Al, m0, Kd, k, lane);
#pragma unroll
    for (int nt = 0; nt < 4; ++nt) {
      v16bf bh = load_b(Wh, n0 + nt * 16, Kd, k, lane);
      v16bf bl = load_b(Wl, n0 + nt * 16, Kd, k, lane);
      c[nt] = bf_wmma(ah, bh, c[nt]);
      c[nt] = bf_wmma(ah, bl, c[nt]);
      c[nt] = bf_wmma(al, bh, c[nt]);
    }
  }

  if (mode == 3) {
    float* o = (float*)outp;
#pragma unroll
    for (int nt = 0; nt < 4; ++nt)
#pragma unroll
      for (int r = 0; r < 8; ++r) {
        int m = m0 + r + hi * 8;
        int n = n0 + nt * 16 + l;
        o[(size_t)m * D_MODEL + n] = c[nt][r] + bias[n];
      }
  } else {
    __bf16* o = (__bf16*)outp;
#pragma unroll
    for (int nt = 0; nt < 4; ++nt)
#pragma unroll
      for (int r = 0; r < 8; ++r) {
        int m = m0 + r + hi * 8;
        int n = n0 + nt * 16 + l;
        int hh = n >> 6, dd = n & 63;
        int bb = m >> 11, tt = m & (SEQ_T - 1);
        size_t idx =
            (mode == 2)
                ? (((size_t)(bb * N_HEADS + hh) * HEAD_DIM + dd) * SEQ_T + tt)
                : (((size_t)(bb * N_HEADS + hh) * SEQ_T + tt) * HEAD_DIM + dd);
        o[idx] = (__bf16)(c[nt][r] + bias[n]);
      }
  }
}

// ---------------- flash attention (causal) ----------------------------------
// Q,K bf16 [BH][T][64]; Vt bf16 [BH][64][T];
// Output: split hi/lo bf16 [B][T][1024] (heads concat) for the final projection.
// grid = (BH=64, T/64=32); block = 128 (4 waves); wave owns 16 query rows.
__global__ __launch_bounds__(128) void attn_fwd_wmma(
    const __bf16* __restrict__ Q, const __bf16* __restrict__ K,
    const __bf16* __restrict__ Vt, __bf16* __restrict__ Oh,
    __bf16* __restrict__ Ol) {
  __shared__ __bf16 lds_p[4][16][32];  // per-wave P staging (C-layout -> A-layout)

  int lane = threadIdx.x & 31;
  int wave = threadIdx.x >> 5;
  int bh = blockIdx.x;
  int q0 = blockIdx.y * 64 + wave * 16;
  int l = lane & 15, hi = lane >> 4;

  const __bf16* Qb = Q + (size_t)bh * SEQ_T * HEAD_DIM;
  const __bf16* Kb = K + (size_t)bh * SEQ_T * HEAD_DIM;
  const __bf16* Vb = Vt + (size_t)bh * HEAD_DIM * SEQ_T;

  v8f acc[4];
#pragma unroll
  for (int nt = 0; nt < 4; ++nt) acc[nt] = (v8f)(0.0f);
  float mrow[8], lrow[8];
#pragma unroll
  for (int r = 0; r < 8; ++r) { mrow[r] = -1e30f; lrow[r] = 0.0f; }

  // Q fragments for this wave's 16 rows, full head dim (k=0 and k=32)
  v16bf qa0 = load_a(Qb, q0, HEAD_DIM, 0, lane);
  v16bf qa1 = load_a(Qb, q0, HEAD_DIM, 32, lane);

  const float scale = 0.125f;  // 1/sqrt(64)
  int kend = q0 + 16;          // causal: keys <= q0+15

  for (int key0 = 0; key0 < kend; key0 += 32) {
    if (key0 + 32 < kend) {  // prefetch next key tile (K rows + one V row/lane)
      __builtin_prefetch(Kb + (size_t)(key0 + 32 + l) * HEAD_DIM, 0, 3);
      __builtin_prefetch(Vb + (size_t)l * SEQ_T + key0 + 32, 0, 3);
    }
    // ---- S = Q K^T for 16x32 score tile (two 16x16 C-fragments) ----
    v8f s[2];
#pragma unroll
    for (int kk = 0; kk < 2; ++kk) {
      v16bf b0 = load_b(Kb, key0 + kk * 16, HEAD_DIM, 0, lane);
      v16bf b1 = load_b(Kb, key0 + kk * 16, HEAD_DIM, 32, lane);
      v8f t = (v8f)(0.0f);
      t = bf_wmma(qa0, b0, t);
      t = bf_wmma(qa1, b1, t);
      s[kk] = t;
    }
    // ---- scale + causal mask ----
#pragma unroll
    for (int kk = 0; kk < 2; ++kk)
#pragma unroll
      for (int r = 0; r < 8; ++r) {
        int key = key0 + kk * 16 + l;
        int qrow = q0 + r + hi * 8;
        float v = s[kk][r] * scale;
        s[kk][r] = (key <= qrow) ? v : -1e30f;
      }
    // ---- online softmax update ----
#pragma unroll
    for (int r = 0; r < 8; ++r) {
      float rm = fmaxf(s[0][r], s[1][r]);
      rm = fmaxf(rm, __shfl_xor(rm, 1));
      rm = fmaxf(rm, __shfl_xor(rm, 2));
      rm = fmaxf(rm, __shfl_xor(rm, 4));
      rm = fmaxf(rm, __shfl_xor(rm, 8));
      float mnew = fmaxf(mrow[r], rm);
      float alpha = __expf(mrow[r] - mnew);
      mrow[r] = mnew;
      float p0 = __expf(s[0][r] - mnew);
      float p1 = __expf(s[1][r] - mnew);
      s[0][r] = p0;
      s[1][r] = p1;
      float rs = p0 + p1;
      rs += __shfl_xor(rs, 1);
      rs += __shfl_xor(rs, 2);
      rs += __shfl_xor(rs, 4);
      rs += __shfl_xor(rs, 8);
      lrow[r] = lrow[r] * alpha + rs;
#pragma unroll
      for (int nt = 0; nt < 4; ++nt) acc[nt][r] *= alpha;
    }
    // ---- stage P (bf16) through per-wave LDS: C-layout -> A-layout ----
#pragma unroll
    for (int r = 0; r < 8; ++r) {
      lds_p[wave][r + hi * 8][l] = (__bf16)s[0][r];
      lds_p[wave][r + hi * 8][16 + l] = (__bf16)s[1][r];
    }
    // same-wave LDS ops are in-order; compiler inserts s_wait_dscnt for uses
    const __bf16* pp = &lds_p[wave][l][hi * 8];
    v8bf p0v = *(const v8bf*)pp;
    v8bf p1v = *(const v8bf*)(pp + 16);
    v16bf pa = cat8(p0v, p1v);
    // ---- acc += P @ V  (Vt rows are head dims; keys contiguous) ----
#pragma unroll
    for (int nt = 0; nt < 4; ++nt) {
      v16bf bv = load_b(Vb, nt * 16, SEQ_T, key0, lane);
      acc[nt] = bf_wmma(pa, bv, acc[nt]);
    }
  }

  // ---- normalize + split-store concat-head bf16 hi/lo [B][T][1024] ----
  int bb = bh >> 4, hh = bh & 15;
#pragma unroll
  for (int nt = 0; nt < 4; ++nt)
#pragma unroll
    for (int r = 0; r < 8; ++r) {
      int t = q0 + r + hi * 8;
      float v = acc[nt][r] / lrow[r];
      size_t idx =
          ((size_t)(bb * SEQ_T + t)) * D_MODEL + hh * HEAD_DIM + nt * 16 + l;
      __bf16 h = (__bf16)v;
      Oh[idx] = h;
      Ol[idx] = (__bf16)(v - (float)h);
    }
}

// ---------------- launcher ---------------------------------------------------

extern "C" void kernel_launch(void* const* d_in, const int* in_sizes, int n_in,
                              void* d_out, int out_size, void* d_ws,
                              size_t ws_size, hipStream_t stream) {
  const float* x  = (const float*)d_in[0];
  // d_in[1] = mask (causal tril) -- implemented analytically, not read
  const float* Wq = (const float*)d_in[2];
  const float* bq = (const float*)d_in[3];
  const float* Wk = (const float*)d_in[4];
  const float* bk = (const float*)d_in[5];
  const float* Wv = (const float*)d_in[6];
  const float* bv = (const float*)d_in[7];
  const float* Wo = (const float*)d_in[8];
  const float* bo = (const float*)d_in[9];

  char* ws = (char*)d_ws;
  const size_t ACT_BYTES = (size_t)M_TOT * D_MODEL * 2;    // 16 MB
  const size_t W_BYTES   = (size_t)D_MODEL * D_MODEL * 2;  // 2 MB

  __bf16* xb_hi = (__bf16*)ws;  ws += ACT_BYTES;
  __bf16* xb_lo = (__bf16*)ws;  ws += ACT_BYTES;
  __bf16* WqH = (__bf16*)ws;    ws += W_BYTES;
  __bf16* WqL = (__bf16*)ws;    ws += W_BYTES;
  __bf16* WkH = (__bf16*)ws;    ws += W_BYTES;
  __bf16* WkL = (__bf16*)ws;    ws += W_BYTES;
  __bf16* WvH = (__bf16*)ws;    ws += W_BYTES;
  __bf16* WvL = (__bf16*)ws;    ws += W_BYTES;
  __bf16* WoH = (__bf16*)ws;    ws += W_BYTES;
  __bf16* WoL = (__bf16*)ws;    ws += W_BYTES;
  __bf16* Qh  = (__bf16*)ws;    ws += ACT_BYTES;
  __bf16* Kh  = (__bf16*)ws;    ws += ACT_BYTES;
  __bf16* Vht = (__bf16*)ws;    ws += ACT_BYTES;
  // Attention output aliases the x cast buffers (x is dead after V projection,
  // and stream ordering serializes the kernels).
  __bf16* attnH = xb_hi;
  __bf16* attnL = xb_lo;

  // 1) casts / weight transposes (split hi+lo)
  {
    int n = M_TOT * D_MODEL;
    cast_split_bf16<<<n / (256 * 4), 256, 0, stream>>>(x, xb_hi, xb_lo, n);
    transpose_split_w<<<1024, 256, 0, stream>>>(Wq, WqH, WqL);
    transpose_split_w<<<1024, 256, 0, stream>>>(Wk, WkH, WkL);
    transpose_split_w<<<1024, 256, 0, stream>>>(Wv, WvH, WvL);
    transpose_split_w<<<1024, 256, 0, stream>>>(Wo, WoH, WoL);
  }

  // 2) Q/K/V projections (3xBF16 split, head-split layouts)
  int gemmGrid = (M_TOT / 128) * (D_MODEL / 64);  // 1024
  gemm_split_wmma<<<gemmGrid, 256, 0, stream>>>(xb_hi, xb_lo, WqH, WqL, bq, Qh,
                                                M_TOT, D_MODEL, 0);
  gemm_split_wmma<<<gemmGrid, 256, 0, stream>>>(xb_hi, xb_lo, WkH, WkL, bk, Kh,
                                                M_TOT, D_MODEL, 0);
  gemm_split_wmma<<<gemmGrid, 256, 0, stream>>>(xb_hi, xb_lo, WvH, WvL, bv, Vht,
                                                M_TOT, D_MODEL, 2);

  // 3) causal flash attention (bf16 WMMA, fp32 online softmax)
  attn_fwd_wmma<<<dim3(BATCH * N_HEADS, SEQ_T / 64), 128, 0, stream>>>(
      Qh, Kh, Vht, attnH, attnL);

  // 4) output projection (3xBF16 split) -> fp32 d_out
  gemm_split_wmma<<<gemmGrid, 256, 0, stream>>>(attnH, attnL, WoH, WoL, bo,
                                                d_out, M_TOT, D_MODEL, 3);
}
